// GCN_755914244220
// MI455X (gfx1250) — matrix operand order repaired and checked by reading, not compiled, not run on hardware
//
#include <hip/hip_runtime.h>
#include <math.h>

typedef __attribute__((ext_vector_type(16))) __bf16 v16bf;
typedef __attribute__((ext_vector_type(8)))  float  v8f;

#define DIN  512
#define DHID 512
#define DOUT 256

// ---------------------------------------------------------------- utilities
__global__ void k_zero(float* p, long long n) {
    long long i = (long long)blockIdx.x * blockDim.x + threadIdx.x;
    long long stride = (long long)gridDim.x * blockDim.x;
    for (; i < n; i += stride) p[i] = 0.0f;
}

__global__ void k_degree(const int* __restrict__ dst, float* __restrict__ deg, int E) {
    int i = blockIdx.x * blockDim.x + threadIdx.x;
    int stride = gridDim.x * blockDim.x;
    for (; i < E; i += stride) atomicAdd(deg + dst[i], 1.0f);
}

__global__ void k_dinv(const float* __restrict__ deg, float* __restrict__ dinv, int n) {
    int i = blockIdx.x * blockDim.x + threadIdx.x;
    int stride = gridDim.x * blockDim.x;
    // +1.0f accounts for the self-loop each node receives
    for (; i < n; i += stride) dinv[i] = rsqrtf(deg[i] + 1.0f);
}

// Pack W[K][Ncol] fp32 (row-major) into Wt[Ncol][K] bf16 (WMMA B-operand layout)
__global__ void k_packT(const float* __restrict__ W, __bf16* __restrict__ Wt, int K, int Ncol) {
    int i = blockIdx.x * blockDim.x + threadIdx.x;
    int stride = gridDim.x * blockDim.x;
    int total = K * Ncol;
    for (; i < total; i += stride) {
        int k = i / Ncol;
        int n = i - k * Ncol;
        Wt[(size_t)n * K + k] = (__bf16)W[i];
    }
}

// ---------------------------------------------------------------- WMMA GEMM
// C[M,Ncol] = A[M,K](fp32, cvt->bf16 inline) x Bt[Ncol,K](bf16, pre-transposed)
// wave tile = 16 rows x 64 cols, 8 waves/block -> 128x64 per block.
// GUARD=false: all 16 rows in range (hot path, no EXEC divergence in k-loop).
template <bool GUARD>
__device__ __forceinline__ void gemm_body(const float* __restrict__ A,
                                          const __bf16* __restrict__ Bt,
                                          float* __restrict__ C,
                                          int M, int K, int Ncol,
                                          int rowBase, int colBase,
                                          int half, int m15) {
    const int arow   = rowBase + m15;
    const bool rowOK = !GUARD || (arow < M);

    v8f acc[4];
    for (int t = 0; t < 4; ++t)
        for (int r = 0; r < 8; ++r) acc[t][r] = 0.0f;

    // A layout: lane holds row m, contiguous 16 K at kBase + 16*half
    const float* aptr = A + (size_t)arow * K + 16 * half;
    // B layout (same k order as A): lane holds col n, contiguous 16 K at kBase + 16*half
    const __bf16* bptr[4];
    for (int t = 0; t < 4; ++t)
        bptr[t] = Bt + (size_t)(colBase + 16 * t + m15) * K + 16 * half;

    for (int k = 0; k < K; k += 32) {
        v16bf a;
        if (rowOK) {
            v8f lo = *(const v8f*)(aptr + k);        // K = k+16h .. +7
            v8f hi = *(const v8f*)(aptr + k + 8);    // K = k+16h+8 .. +15
            for (int i = 0; i < 8; ++i) {
                a[i]     = (__bf16)lo[i];
                a[i + 8] = (__bf16)hi[i];
            }
        } else {
            for (int i = 0; i < 16; ++i) a[i] = (__bf16)0.0f;
        }
        for (int t = 0; t < 4; ++t) {
            v16bf b = *(const v16bf*)(bptr[t] + k);  // 32B contiguous
            acc[t] = __builtin_amdgcn_wmma_f32_16x16x32_bf16(
                false, a, false, b, (short)0, acc[t], false, false);
        }
    }

    // C/D layout: VGPR r -> row r + 8*half, col = lane&15
    for (int t = 0; t < 4; ++t) {
        const int col = colBase + 16 * t + m15;
        for (int r = 0; r < 8; ++r) {
            const int orow = rowBase + r + 8 * half;
            if (!GUARD || orow < M)
                C[(size_t)orow * Ncol + col] = acc[t][r];
        }
    }
}

__global__ __launch_bounds__(256)
void k_gemm_wmma(const float* __restrict__ A, const __bf16* __restrict__ Bt,
                 float* __restrict__ C, int M, int K, int Ncol) {
    const int lane = threadIdx.x & 31;
    const int wave = threadIdx.x >> 5;      // 0..7
    const int half = lane >> 4;             // 0 or 1
    const int m15  = lane & 15;

    // grid.x = column blocks (fast dim) so all col-blocks of the same 128 rows
    // run adjacently -> A tile (256KB) stays L2-resident, A streamed ~once.
    const int colBase = blockIdx.x * 64;
    const int rowBase = (blockIdx.y * 8 + wave) * 16;

    if (rowBase + 16 <= M) {
        gemm_body<false>(A, Bt, C, M, K, Ncol, rowBase, colBase, half, m15);
    } else if (rowBase < M) {
        gemm_body<true>(A, Bt, C, M, K, Ncol, rowBase, colBase, half, m15);
    }
}

// ---------------------------------------------------------------- scatter-add
// out[dst] += h[src] * dinv[src]*dinv[dst]  over E edges + N self-loops
__global__ __launch_bounds__(256)
void k_scatter(const float* __restrict__ h, const int* __restrict__ src,
               const int* __restrict__ dst, const float* __restrict__ dinv,
               float* __restrict__ out, int D, int E, int Nn) {
    const int total = E + Nn;
    for (int e = blockIdx.x; e < total; e += gridDim.x) {
        int s, d;
        if (e < E) { s = src[e]; d = dst[e]; }
        else       { s = e - E;  d = s; }
        const float nrm = dinv[s] * dinv[d];
        const float* hp = h + (size_t)s * D;
        float* op = out + (size_t)d * D;
        for (int j = threadIdx.x; j < D; j += blockDim.x)
            atomicAdd(op + j, hp[j] * nrm);
    }
}

// in-place: S = relu(S + b)
__global__ void k_bias_relu(float* __restrict__ S, const float* __restrict__ b,
                            long long total, int D) {
    long long i = (long long)blockIdx.x * blockDim.x + threadIdx.x;
    long long stride = (long long)gridDim.x * blockDim.x;
    for (; i < total; i += stride) {
        int c = (int)(i & (long long)(D - 1));     // D is a power of 2
        S[i] = fmaxf(S[i] + b[c], 0.0f);
    }
}

// per-row: out = (out + b) - logsumexp(out + b)   (D == blockDim.x == 256)
__global__ __launch_bounds__(256)
void k_logsoftmax(float* __restrict__ out, const float* __restrict__ bias, int D) {
    __shared__ float red[256];
    const int row = blockIdx.x;
    float* p = out + (size_t)row * D;
    const int t = threadIdx.x;
    float v = p[t] + bias[t];

    red[t] = v; __syncthreads();
    for (int s = 128; s > 0; s >>= 1) {
        if (t < s) red[t] = fmaxf(red[t], red[t + s]);
        __syncthreads();
    }
    const float mx = red[0]; __syncthreads();

    red[t] = expf(v - mx); __syncthreads();
    for (int s = 128; s > 0; s >>= 1) {
        if (t < s) red[t] += red[t + s];
        __syncthreads();
    }
    const float lse = mx + logf(red[0]);
    p[t] = v - lse;
}

// ---------------------------------------------------------------- launcher
extern "C" void kernel_launch(void* const* d_in, const int* in_sizes, int n_in,
                              void* d_out, int out_size, void* d_ws, size_t ws_size,
                              hipStream_t stream) {
    (void)n_in; (void)out_size; (void)ws_size;
    const float* x  = (const float*)d_in[0];
    const int*   ei = (const int*)  d_in[1];
    const float* W1 = (const float*)d_in[2];
    const float* b1 = (const float*)d_in[3];
    const float* W2 = (const float*)d_in[4];
    const float* b2 = (const float*)d_in[5];
    float* out = (float*)d_out;

    const int N = in_sizes[0] / DIN;
    const int E = in_sizes[1] / 2;
    const int* src = ei;
    const int* dst = ei + E;

    char* ws = (char*)d_ws;
    size_t off = 0;
    auto alloc = [&](size_t bytes) -> char* {
        char* p = ws + off;
        off = (off + bytes + 255) & ~(size_t)255;
        return p;
    };
    float*  deg  = (float*) alloc((size_t)N * 4);
    float*  dinv = (float*) alloc((size_t)N * 4);
    __bf16* W1t  = (__bf16*)alloc((size_t)DIN  * DHID * 2);
    __bf16* W2t  = (__bf16*)alloc((size_t)DHID * DOUT * 2);
    float*  H1   = (float*) alloc((size_t)N * DHID * 4);  // reused as H2
    float*  S1   = (float*) alloc((size_t)N * DHID * 4);
    float*  H2   = H1;

    // zero-init (harness does not clear between replays)
    k_zero<<<2048, 256, 0, stream>>>(deg, (long long)N);
    k_zero<<<8192, 256, 0, stream>>>(S1,  (long long)N * DHID);
    k_zero<<<8192, 256, 0, stream>>>(out, (long long)N * DOUT);

    // normalization
    k_degree<<<2048, 256, 0, stream>>>(dst, deg, E);
    k_dinv  <<<1024, 256, 0, stream>>>(deg, dinv, N);

    // pack weights bf16, transposed to B-operand layout
    k_packT<<<1024, 256, 0, stream>>>(W1, W1t, DIN,  DHID);
    k_packT<<<512,  256, 0, stream>>>(W2, W2t, DHID, DOUT);

    // layer 1: H1 = X @ W1 ; S1 = scatter(H1) ; S1 = relu(S1 + b1)
    dim3 g1(DHID / 64, (N + 127) / 128);
    k_gemm_wmma<<<g1, 256, 0, stream>>>(x, W1t, H1, N, DIN, DHID);
    k_scatter  <<<8192, 256, 0, stream>>>(H1, src, dst, dinv, S1, DHID, E, N);
    k_bias_relu<<<8192, 256, 0, stream>>>(S1, b1, (long long)N * DHID, DHID);

    // layer 2: H2 = S1 @ W2 ; out = scatter(H2) ; out = log_softmax(out + b2)
    dim3 g2(DOUT / 64, (N + 127) / 128);
    k_gemm_wmma<<<g2, 256, 0, stream>>>(S1, W2t, H2, N, DHID, DOUT);
    k_scatter  <<<8192, 256, 0, stream>>>(H2, src, dst, dinv, out, DOUT, E, N);
    k_logsoftmax<<<N, 256, 0, stream>>>(out, b2, DOUT);
}
